// PointNetPPClassifier_65953517797453
// MI455X (gfx1250) — matrix operand order repaired and checked by reading, not compiled, not run on hardware
//
#include <hip/hip_runtime.h>

// ---------------------------------------------------------------------------
// PointNet++ MSG forward for MI455X (gfx1250, wave32).
// Hot path: v_wmma_f32_16x16x32_f16 GEMMs. Each wave computes a 16x32 output
// tile (2 WMMAs / K-step sharing one A fragment); 4 waves/block => 64x32 per
// block. A/B tiles staged in LDS as f16 with branch-free vectorized fills.
// ---------------------------------------------------------------------------

typedef __attribute__((ext_vector_type(16))) _Float16 v16h;
typedef __attribute__((ext_vector_type(8)))  _Float16 v8h;
typedef __attribute__((ext_vector_type(8)))  float    v8f;

#define BATCH 16
#define NPTS  4096

// ----------------------------- glue kernels --------------------------------

__global__ void split_xyz(const float* __restrict__ xyz, float* __restrict__ coords,
                          float* __restrict__ nrm, long long total /* B*N */) {
  long long i = blockIdx.x * (long long)blockDim.x + threadIdx.x;
  long long stride = (long long)gridDim.x * blockDim.x;
  for (; i < total; i += stride) {
    coords[i * 3 + 0] = xyz[i * 6 + 0];
    coords[i * 3 + 1] = xyz[i * 6 + 1];
    coords[i * 3 + 2] = xyz[i * 6 + 2];
    nrm[i * 3 + 0]    = xyz[i * 6 + 3];
    nrm[i * 3 + 1]    = xyz[i * 6 + 4];
    nrm[i * 3 + 2]    = xyz[i * 6 + 5];
  }
}

// Farthest-point sampling: one workgroup per batch, distances live in LDS.
__global__ void fps_kernel(const float* __restrict__ coords, int n, int npoint,
                           int* __restrict__ out_idx) {
  __shared__ float dist[4096];
  __shared__ float rv[256];
  __shared__ int   ri[256];
  __shared__ int   sfar;
  const int b = blockIdx.x, t = threadIdx.x, T = blockDim.x;
  const float* pts = coords + (long long)b * n * 3;
  for (int i = t; i < n; i += T) dist[i] = 1e10f;
  if (t == 0) sfar = 0;
  __syncthreads();
  for (int s = 0; s < npoint; ++s) {
    const int far = sfar;
    if (t == 0) out_idx[b * npoint + s] = far;
    const float fx = pts[far * 3 + 0], fy = pts[far * 3 + 1], fz = pts[far * 3 + 2];
    float best = -1.0f; int bi = 0;
    for (int i = t; i < n; i += T) {
      float dx = pts[i * 3 + 0] - fx, dy = pts[i * 3 + 1] - fy, dz = pts[i * 3 + 2] - fz;
      float nd = fminf(dist[i], dx * dx + dy * dy + dz * dz);
      dist[i] = nd;
      if (nd > best) { best = nd; bi = i; }
    }
    rv[t] = best; ri[t] = bi;
    __syncthreads();
    for (int o = T >> 1; o > 0; o >>= 1) {
      if (t < o) {
        if (rv[t + o] > rv[t] || (rv[t + o] == rv[t] && ri[t + o] < ri[t])) {
          rv[t] = rv[t + o]; ri[t] = ri[t + o];
        }
      }
      __syncthreads();
    }
    if (t == 0) sfar = ri[0];
    __syncthreads();
  }
}

__global__ void gather3(const float* __restrict__ src, const int* __restrict__ idx,
                        int n, int np, float* __restrict__ dst, long long total /*B*np*3*/) {
  long long i = blockIdx.x * (long long)blockDim.x + threadIdx.x;
  long long stride = (long long)gridDim.x * blockDim.x;
  for (; i < total; i += stride) {
    int d = (int)(i % 3);
    long long rem = i / 3;
    int s = (int)(rem % np);
    int b = (int)(rem / np);
    dst[i] = src[((long long)b * n + idx[b * np + s]) * 3 + d];
  }
}

// Reference's sort(where(sqr>r2, n, arange))[:k] == first k in-radius indices
// in increasing order, padded with the first hit.
__global__ void ball_query(const float* __restrict__ xyz, const float* __restrict__ centers,
                           int n, int np, float r2, int k, int* __restrict__ idx,
                           int total /* B*np */) {
  int gid = blockIdx.x * blockDim.x + threadIdx.x;
  if (gid >= total) return;
  int b = gid / np;
  const float* p = xyz + (long long)b * n * 3;
  const float cx = centers[gid * 3 + 0], cy = centers[gid * 3 + 1], cz = centers[gid * 3 + 2];
  int* o = idx + (long long)gid * k;
  int cnt = 0, first = 0;
  for (int j = 0; j < n && cnt < k; ++j) {
    float dx = p[j * 3 + 0] - cx, dy = p[j * 3 + 1] - cy, dz = p[j * 3 + 2] - cz;
    if (dx * dx + dy * dy + dz * dz <= r2) {
      if (cnt == 0) first = j;
      o[cnt++] = j;
    }
  }
  for (; cnt < k; ++cnt) o[cnt] = first;
}

// feat[row, :] = [ points[b, idx, 0..pc), coords[b, idx, :]-center[b, s, :] ]
__global__ void group_feat(const float* __restrict__ coords, const float* __restrict__ points,
                           int pc, const float* __restrict__ centers, const int* __restrict__ idx,
                           int n, int np, int k, float* __restrict__ feat, long long total) {
  const int C = pc + 3;
  long long i = blockIdx.x * (long long)blockDim.x + threadIdx.x;
  long long stride = (long long)gridDim.x * blockDim.x;
  for (; i < total; i += stride) {
    long long row = i / C;
    int c = (int)(i % C);
    int b = (int)(row / ((long long)np * k));
    long long rem = row % ((long long)np * k);
    int s = (int)(rem / k);
    int pj = idx[row];
    float v;
    if (c < pc) v = points[((long long)b * n + pj) * pc + c];
    else {
      int d = c - pc;
      v = coords[((long long)b * n + pj) * 3 + d] - centers[((long long)b * np + s) * 3 + d];
    }
    feat[i] = v;
  }
}

// ------------------------------ WMMA GEMM ----------------------------------
// Y[rows,cout] = X[rows,cin] * W[cout,cin]^T + bias.
// Block = 128 threads (4 waves). Wave w: rows [rb, rb+16), cols [ob, ob+32).
// LDS B layout [n][k] == native W[cout][cin]; A layout [m][k]. Fills are
// branch-free: guards hoisted to wave/block-uniform fast/slow path selection.
__global__ void gemm_bias_wmma(const float* __restrict__ X, const float* __restrict__ W,
                               const float* __restrict__ bias, float* __restrict__ Y,
                               long long rows, int cin, int cout) {
  __shared__ _Float16 ldsB[32][32];     // [n][k]
  __shared__ _Float16 ldsA[4][16][32];  // per-wave [m][k]
  const int tid = threadIdx.x;
  const int wave = tid >> 5, lane = tid & 31;
  const long long rb = (long long)blockIdx.x * 64 + wave * 16;
  const int ob = blockIdx.y * 32;
  const int mh = lane & 15, hi = lane >> 4;
  const bool rowFull = (rb + 16 <= rows);
  // Fill ownership: A -> lane covers row ar, halves [ac, ac+16); B -> thread
  // covers row bn, halves [bc, bc+8). Contiguous => vector LDS stores.
  const int ar = lane >> 1, ac = (lane & 1) * 16;
  const int bn = tid >> 2, bc = (tid & 3) * 8;
  const long long arow = (rb + ar < rows ? rb + ar : rows - 1);
  const float* __restrict__ xrow = X + arow * cin;
  const float* __restrict__ wrow = W + (long long)(ob + bn) * cin;

  v8f acc0 = {}, acc1 = {};
  for (int k0 = 0; k0 < cin; k0 += 32) {
    const bool fullK = (k0 + 32 <= cin);
    __syncthreads();
    // ---- B tile fill ----
    if (fullK) {
      const float* ws = wrow + k0 + bc;
      v8h tb;
#pragma unroll
      for (int e = 0; e < 8; ++e) tb[e] = (_Float16)ws[e];
      *(v8h*)&ldsB[bn][bc] = tb;
    } else {
      v8h tb;
#pragma unroll
      for (int e = 0; e < 8; ++e) {
        int kk = k0 + bc + e;
        float v = wrow[kk < cin ? kk : cin - 1];
        tb[e] = (_Float16)(kk < cin ? v : 0.0f);
      }
      *(v8h*)&ldsB[bn][bc] = tb;
    }
    // ---- A tile fill (per wave) ----
    if (fullK && rowFull) {
      const float* xs = xrow + k0 + ac;
      __builtin_prefetch((const void*)(xs + 32), 0, 0);  // next K block
      v16h ta;
#pragma unroll
      for (int e = 0; e < 16; ++e) ta[e] = (_Float16)xs[e];
      *(v16h*)&ldsA[wave][ar][ac] = ta;
    } else {
      const bool rok = (rb + ar < rows);
      v16h ta;
#pragma unroll
      for (int e = 0; e < 16; ++e) {
        int kk = k0 + ac + e;
        float v = xrow[kk < cin ? kk : cin - 1];
        ta[e] = (_Float16)((rok && kk < cin) ? v : 0.0f);
      }
      *(v16h*)&ldsA[wave][ar][ac] = ta;
    }
    __syncthreads();
    // ---- fragment pack (contiguous K = hi*16 + e in A and B) + 2x WMMA ----
    v16h a, b0, b1;
#pragma unroll
    for (int e = 0; e < 8; ++e) {
      a[e]     = ldsA[wave][mh][hi * 16 + e];       // K = hi*16 + e
      a[8 + e] = ldsA[wave][mh][hi * 16 + 8 + e];   // K = hi*16 + 8 + e
    }
#pragma unroll
    for (int e = 0; e < 16; ++e) {
      b0[e] = ldsB[mh][hi * 16 + e];                // N tile 0
      b1[e] = ldsB[16 + mh][hi * 16 + e];           // N tile 1
    }
    acc0 = __builtin_amdgcn_wmma_f32_16x16x32_f16(false, a, false, b0,
                                                  (short)0, acc0, false, false);
    acc1 = __builtin_amdgcn_wmma_f32_16x16x32_f16(false, a, false, b1,
                                                  (short)0, acc1, false, false);
  }
  // C/D layout: VGPR r -> M = hi*8 + r, N = mh.
  const float bias0 = bias[ob + mh], bias1 = bias[ob + 16 + mh];
  if (rowFull) {
#pragma unroll
    for (int r8 = 0; r8 < 8; ++r8) {
      long long row = rb + hi * 8 + r8;
      Y[row * cout + ob + mh]      = acc0[r8] + bias0;
      Y[row * cout + ob + 16 + mh] = acc1[r8] + bias1;
    }
  } else {
#pragma unroll
    for (int r8 = 0; r8 < 8; ++r8) {
      long long row = rb + hi * 8 + r8;
      if (row < rows) {
        Y[row * cout + ob + mh]      = acc0[r8] + bias0;
        Y[row * cout + ob + 16 + mh] = acc1[r8] + bias1;
      }
    }
  }
}

// ----------------------- BatchNorm (over all rows) -------------------------

__global__ void bn_stats(const float* __restrict__ Y, long long rows, int cout,
                         const float* __restrict__ g, const float* __restrict__ be,
                         float* __restrict__ scale, float* __restrict__ shift) {
  const int c = blockIdx.x, t = threadIdx.x;
  __shared__ float s1[256], s2[256];
  float a = 0.0f, b2 = 0.0f;
  for (long long r = t; r < rows; r += blockDim.x) {
    float v = Y[r * cout + c];
    a += v; b2 += v * v;
  }
  s1[t] = a; s2[t] = b2;
  __syncthreads();
  for (int o = 128; o > 0; o >>= 1) {
    if (t < o) { s1[t] += s1[t + o]; s2[t] += s2[t + o]; }
    __syncthreads();
  }
  if (t == 0) {
    float inv = 1.0f / (float)rows;
    float m = s1[0] * inv;
    float v = s2[0] * inv - m * m;
    float sc = g[c] * rsqrtf(v + 1e-5f);
    scale[c] = sc;
    shift[c] = be[c] - m * sc;
  }
}

__global__ void bn_relu(float* __restrict__ Y, long long total, int cout,
                        const float* __restrict__ scale, const float* __restrict__ shift) {
  long long i = blockIdx.x * (long long)blockDim.x + threadIdx.x;
  long long stride = (long long)gridDim.x * blockDim.x;
  for (; i < total; i += stride) {
    int c = (int)(i % cout);
    Y[i] = fmaxf(Y[i] * scale[c] + shift[c], 0.0f);
  }
}

__global__ void maxpool_k(const float* __restrict__ feat, int k, int cout,
                          float* __restrict__ out, int ostride, int coff,
                          long long nrows) {
  long long total = nrows * cout;
  long long i = blockIdx.x * (long long)blockDim.x + threadIdx.x;
  long long stride = (long long)gridDim.x * blockDim.x;
  for (; i < total; i += stride) {
    long long row = i / cout;
    int c = (int)(i % cout);
    float m = -3.4e38f;
    for (int j = 0; j < k; ++j) m = fmaxf(m, feat[(row * k + j) * (long long)cout + c]);
    out[row * ostride + coff + c] = m;
  }
}

__global__ void concat_feat(const float* __restrict__ xyz3, const float* __restrict__ pts,
                            int pc, float* __restrict__ out, long long total) {
  const int C = 3 + pc;
  long long i = blockIdx.x * (long long)blockDim.x + threadIdx.x;
  long long stride = (long long)gridDim.x * blockDim.x;
  for (; i < total; i += stride) {
    long long row = i / C;
    int c = (int)(i % C);
    out[i] = (c < 3) ? xyz3[row * 3 + c] : pts[row * pc + (c - 3)];
  }
}

__global__ void final_reduce(const float* __restrict__ x, const float* __restrict__ coef,
                             float* __restrict__ out) {
  const int b = blockIdx.x, t = threadIdx.x;
  __shared__ float s[256];
  float a = 0.0f;
  for (int c = t; c < 512; c += blockDim.x) {
    float v = x[b * 512 + c];
    v = 1.0f / (1.0f + expf(-v));
    v *= coef[b * 512 + c];
    a += fmaxf(v, 0.0f);
  }
  s[t] = a;
  __syncthreads();
  for (int o = 128; o > 0; o >>= 1) {
    if (t < o) s[t] += s[t + o];
    __syncthreads();
  }
  if (t == 0) out[b] = s[0];
}

// ------------------------------- host side ---------------------------------

struct LayerP { const float *W, *b, *g, *be; int cin, cout; };

static inline unsigned gridFor(long long n, int tpb, unsigned cap = 8192u) {
  long long blk = (n + tpb - 1) / tpb;
  if (blk < 1) blk = 1;
  if (blk > (long long)cap) blk = cap;
  return (unsigned)blk;
}

static void run_conv(const float* X, float* Y, long long rows, const LayerP& L,
                     float* scale, float* shift, hipStream_t s) {
  dim3 grid((unsigned)((rows + 63) / 64), (unsigned)(L.cout / 32));
  gemm_bias_wmma<<<grid, 128, 0, s>>>(X, L.W, L.b, Y, rows, L.cin, L.cout);
  bn_stats<<<L.cout, 256, 0, s>>>(Y, rows, L.cout, L.g, L.be, scale, shift);
  long long total = rows * L.cout;
  bn_relu<<<gridFor(total, 256), 256, 0, s>>>(Y, total, L.cout, scale, shift);
}

extern "C" void kernel_launch(void* const* d_in, const int* in_sizes, int n_in,
                              void* d_out, int out_size, void* d_ws, size_t ws_size,
                              hipStream_t stream) {
  (void)in_sizes; (void)n_in; (void)out_size; (void)ws_size;
  const float* xyz  = (const float*)d_in[0];
  const float* coef = (const float*)d_in[1];

  // Parameter order follows setup_inputs() insertion order:
  // sa1(3 branches x 3 layers), sa2, sa3, fc1..fc3; each dense = W,b,g,be.
  int p = 2;
  auto nextL = [&](int cin, int cout) {
    LayerP L{(const float*)d_in[p], (const float*)d_in[p + 1],
             (const float*)d_in[p + 2], (const float*)d_in[p + 3], cin, cout};
    p += 4;
    return L;
  };
  const int sa1c[3][4] = {{6, 32, 32, 64}, {6, 64, 64, 128}, {6, 64, 96, 128}};
  const int sa2c[3][4] = {{323, 64, 64, 128}, {323, 128, 128, 256}, {323, 128, 128, 256}};
  const int sa3c[4] = {643, 256, 512, 1024};
  LayerP sa1[3][3], sa2[3][3], sa3[3];
  for (int br = 0; br < 3; ++br)
    for (int i = 0; i < 3; ++i) sa1[br][i] = nextL(sa1c[br][i], sa1c[br][i + 1]);
  for (int br = 0; br < 3; ++br)
    for (int i = 0; i < 3; ++i) sa2[br][i] = nextL(sa2c[br][i], sa2c[br][i + 1]);
  for (int i = 0; i < 3; ++i) sa3[i] = nextL(sa3c[i], sa3c[i + 1]);
  LayerP fc1 = nextL(1024, 512), fc2 = nextL(512, 256), fc3 = nextL(256, 512);

  // Workspace layout (floats). Ping-pong buffers cover the largest layer:
  // sa1 k=128 branch -> 16*512*128 rows x 128 ch.
  float* f = (float*)d_ws;
  size_t o = 0;
  auto alloc = [&](size_t n) { float* r = f + o; o += (n + 63) & ~(size_t)63; return r; };
  const size_t BIG = (size_t)BATCH * 512 * 128 * 128;  // 134,217,728 floats
  float* bufA   = alloc(BIG);
  float* bufB   = alloc(BIG);
  float* coords = alloc((size_t)BATCH * NPTS * 3);
  float* nrm    = alloc((size_t)BATCH * NPTS * 3);
  float* l1xyz  = alloc((size_t)BATCH * 512 * 3);
  float* l2xyz  = alloc((size_t)BATCH * 128 * 3);
  float* l1pts  = alloc((size_t)BATCH * 512 * 320);
  float* l2pts  = alloc((size_t)BATCH * 128 * 640);
  float* fcA    = alloc((size_t)BATCH * 1024);
  float* fcB    = alloc((size_t)BATCH * 1024);
  float* scale  = alloc(1024);
  float* shift  = alloc(1024);
  int*   fpsidx  = (int*)alloc((size_t)BATCH * 512);
  int*   ballidx = (int*)alloc((size_t)BATCH * 512 * 128);

  // ---- split xyz -> coords, norm ----
  split_xyz<<<gridFor((long long)BATCH * NPTS, 256), 256, 0, stream>>>(
      xyz, coords, nrm, (long long)BATCH * NPTS);

  // ---- SA1: 512 centers, radii {0.1,0.2,0.4}, k {16,32,128}, points=norm ----
  fps_kernel<<<BATCH, 256, 0, stream>>>(coords, NPTS, 512, fpsidx);
  gather3<<<gridFor((long long)BATCH * 512 * 3, 256), 256, 0, stream>>>(
      coords, fpsidx, NPTS, 512, l1xyz, (long long)BATCH * 512 * 3);
  {
    const float rr[3] = {0.1f, 0.2f, 0.4f};
    const int kk[3] = {16, 32, 128};
    int off = 0;
    for (int br = 0; br < 3; ++br) {
      const int k = kk[br];
      const long long rows = (long long)BATCH * 512 * k;
      ball_query<<<gridFor(BATCH * 512, 128), 128, 0, stream>>>(
          coords, l1xyz, NPTS, 512, rr[br] * rr[br], k, ballidx, BATCH * 512);
      group_feat<<<gridFor(rows * 6, 256), 256, 0, stream>>>(
          coords, nrm, 3, l1xyz, ballidx, NPTS, 512, k, bufA, rows * 6);
      const float* cur = bufA; float* nxt = bufB;
      for (int li = 0; li < 3; ++li) {
        run_conv(cur, nxt, rows, sa1[br][li], scale, shift, stream);
        cur = nxt; nxt = (nxt == bufB) ? bufA : bufB;
      }
      const int co = sa1[br][2].cout;
      maxpool_k<<<gridFor((long long)BATCH * 512 * co, 256), 256, 0, stream>>>(
          cur, k, co, l1pts, 320, off, (long long)BATCH * 512);
      off += co;
    }
  }

  // ---- SA2: 128 centers, radii {0.2,0.4,0.8}, k {32,64,128}, points=l1pts ----
  fps_kernel<<<BATCH, 256, 0, stream>>>(l1xyz, 512, 128, fpsidx);
  gather3<<<gridFor((long long)BATCH * 128 * 3, 256), 256, 0, stream>>>(
      l1xyz, fpsidx, 512, 128, l2xyz, (long long)BATCH * 128 * 3);
  {
    const float rr[3] = {0.2f, 0.4f, 0.8f};
    const int kk[3] = {32, 64, 128};
    int off = 0;
    for (int br = 0; br < 3; ++br) {
      const int k = kk[br];
      const long long rows = (long long)BATCH * 128 * k;
      ball_query<<<gridFor(BATCH * 128, 128), 128, 0, stream>>>(
          l1xyz, l2xyz, 512, 128, rr[br] * rr[br], k, ballidx, BATCH * 128);
      group_feat<<<gridFor(rows * 323, 256), 256, 0, stream>>>(
          l1xyz, l1pts, 320, l2xyz, ballidx, 512, 128, k, bufA, rows * 323);
      const float* cur = bufA; float* nxt = bufB;
      for (int li = 0; li < 3; ++li) {
        run_conv(cur, nxt, rows, sa2[br][li], scale, shift, stream);
        cur = nxt; nxt = (nxt == bufB) ? bufA : bufB;
      }
      const int co = sa2[br][2].cout;
      maxpool_k<<<gridFor((long long)BATCH * 128 * co, 256), 256, 0, stream>>>(
          cur, k, co, l2pts, 640, off, (long long)BATCH * 128);
      off += co;
    }
  }

  // ---- SA3 (group-all style): feat = [l2_xyz, l2_pts], rows = B*128 ----
  {
    const long long rows = (long long)BATCH * 128;
    concat_feat<<<gridFor(rows * 643, 256), 256, 0, stream>>>(
        l2xyz, l2pts, 640, bufA, rows * 643);
    const float* cur = bufA; float* nxt = bufB;
    for (int li = 0; li < 3; ++li) {
      run_conv(cur, nxt, rows, sa3[li], scale, shift, stream);
      cur = nxt; nxt = (nxt == bufB) ? bufA : bufB;
    }
    // max over the 128 grouped points -> (B, 1024)
    maxpool_k<<<gridFor((long long)BATCH * 1024, 256), 256, 0, stream>>>(
        cur, 128, 1024, fcA, 1024, 0, BATCH);
  }

  // ---- FC head ----
  run_conv(fcA, fcB, BATCH, fc1, scale, shift, stream);   // (16,512)
  run_conv(fcB, fcA, BATCH, fc2, scale, shift, stream);   // (16,256)
  {
    dim3 g3(1, 512 / 32);                                 // fc3: no BN
    gemm_bias_wmma<<<g3, 128, 0, stream>>>(fcA, fc3.W, fc3.b, fcB,
                                           (long long)BATCH, 256, 512);
  }
  final_reduce<<<BATCH, 256, 0, stream>>>(fcB, coef, (float*)d_out);
}
